// AutoCorr2D_18700287607587
// MI455X (gfx1250) — matrix operand order repaired and checked by reading, not compiled, not run on hardware
//
#include <hip/hip_runtime.h>

#define BATCH 8
#define CIN   256
#define HH    64
#define WW    64
#define CCH   128
#define COUTC 64

typedef __attribute__((ext_vector_type(16))) _Float16 v16h;
typedef __attribute__((ext_vector_type(2)))  _Float16 h2;
typedef __attribute__((ext_vector_type(8)))  float    v8f;

union U16x16 { v16h v; unsigned int u[8]; };
union U32h   { unsigned int u; h2 h; unsigned short b[2]; };

__device__ __forceinline__ unsigned short f32_to_f16b(float x) {
  union { _Float16 h; unsigned short u; } c;
  c.h = (_Float16)x;
  return c.u;
}

// Async copy of 16 bytes global -> LDS (CDNA5 Tensor/async path, ASYNCcnt).
__device__ __forceinline__ void async_g2l_b128(unsigned lds_off,
                                               const unsigned int* gp) {
  asm volatile("global_load_async_to_lds_b128 %0, %1, off"
               :: "v"(lds_off), "v"(gp) : "memory");
}
__device__ __forceinline__ void wait_asynccnt0() {
  asm volatile("s_wait_asynccnt 0x0" ::: "memory");
}

// ---------------------------------------------------------------------------
// Prep: one-time f32->f16 conversions.
//   w_ext (CC,CIN,3,3)  ->  wext16[kv][cc][cin]      (A rows contiguous in K)
//   w_reg (COUT,CC,5,5) ->  wreg16[k][o][c]
//   x     (B,CIN,H,W)   ->  xp[(b*128+cp)][i][j]     packed f16 channel pairs
// ---------------------------------------------------------------------------
__global__ void prep_convert(const float* __restrict__ wext,
                             const float* __restrict__ wreg,
                             const float* __restrict__ x,
                             unsigned short* __restrict__ wext16,
                             unsigned short* __restrict__ wreg16,
                             unsigned int* __restrict__ xp) {
  int tid    = blockIdx.x * blockDim.x + threadIdx.x;
  int stride = gridDim.x * blockDim.x;
  for (int i = tid; i < 9 * CCH * CIN; i += stride) {
    int kv = i / (CCH * CIN);
    int r  = i % (CCH * CIN);
    int cc = r / CIN;
    int ci = r % CIN;
    wext16[i] = f32_to_f16b(wext[(cc * CIN + ci) * 9 + kv]);
  }
  for (int i = tid; i < 25 * COUTC * CCH; i += stride) {
    int k = i / (COUTC * CCH);
    int r = i % (COUTC * CCH);
    int o = r / CCH;
    int c = r % CCH;
    wreg16[i] = f32_to_f16b(wreg[(o * CCH + c) * 25 + k]);
  }
  const int NXP = BATCH * (CIN / 2) * HH * WW;   // 4.19M dwords
  for (int d = tid; d < NXP; d += stride) {
    int b   = d / ((CIN / 2) * HH * WW);
    int r   = d % ((CIN / 2) * HH * WW);
    int cp  = r / (HH * WW);
    int pix = r % (HH * WW);
    const float* p = x + ((size_t)b * CIN + 2 * cp) * (HH * WW) + pix;
    U32h u;
    u.b[0] = f32_to_f16b(p[0]);
    u.b[1] = f32_to_f16b(p[HH * WW]);
    xp[d] = u.u;
  }
}

// ---------------------------------------------------------------------------
// Stage 1: 3x3 conv CIN=256 -> CC=128, implicit GEMM on v_wmma_f32_16x16x32_f16.
// One workgroup per (b,row). LDS tile [cp=128][3 rows][72 cols], interior cols
// 4..67 filled by global_load_async_to_lds_b128, halos zeroed with ds_store_b128.
// ---------------------------------------------------------------------------
#define XCOLS 72
__global__ void __launch_bounds__(256)
conv3x3_wmma(const unsigned int* __restrict__ xp,
             const unsigned short* __restrict__ wext16,
             const float* __restrict__ bext,
             unsigned int* __restrict__ fws) {
  __shared__ unsigned int lds[(CIN / 2) * 3 * XCOLS]; // 27648 dwords = 108 KB

  const int wg = blockIdx.x;
  const int b  = wg >> 6;
  const int i  = wg & 63;
  const int tid = threadIdx.x;

  const uint4 z4 = {0u, 0u, 0u, 0u};
  const int NSEG = (CIN / 2) * 3;                 // 384 (cp, row) segments
  // Halo columns (gj = -4..-1 and 64..67): one b128 each side.
  for (int s = tid; s < NSEG; s += 256) {
    *(uint4*)&lds[s * XCOLS + 0]  = z4;
    *(uint4*)&lds[s * XCOLS + 68] = z4;
  }
  // Interior: async-copy 64 dwords per segment (16B per lane-op).
  for (int w = tid; w < NSEG * 16; w += 256) {
    int s  = w >> 4, q = w & 15;
    int cp = s / 3, r = s % 3;
    int gi = i + r - 1;
    int lidx = s * XCOLS + 4 + q * 4;
    if (gi >= 0 && gi < HH) {
      const unsigned int* gp =
          xp + ((((size_t)b * (CIN / 2) + cp) * HH + gi) * WW + q * 4);
      async_g2l_b128((unsigned)(uintptr_t)&lds[lidx], gp);
    } else {
      *(uint4*)&lds[lidx] = z4;
    }
  }
  wait_asynccnt0();
  __syncthreads();

  const int lane   = tid & 31;
  const int wv     = tid >> 5;
  const int lane15 = lane & 15;
  const int hi     = (lane >> 4) & 1;
  const int nt     = wv & 3;       // pixel tile
  const int mhalf  = wv >> 2;      // CC half
  const int px     = nt * 16 + lane15;

  v8f acc[4] = {};

  for (int kv = 0; kv < 9; ++kv) {
    const int du = kv / 3, dv = kv % 3;
    const int col = px + dv + 3;             // LDS col: gj = j+dv-1 -> +4
    for (int cb = 0; cb < 8; ++cb) {         // 8 blocks of 32 cin
      U16x16 Bm;
#pragma unroll
      for (int jv = 0; jv < 8; ++jv) {
        int cp = cb * 16 + jv + hi * 8;
        Bm.u[jv] = lds[(cp * 3 + du) * XCOLS + col];
      }
#pragma unroll
      for (int mt = 0; mt < 4; ++mt) {
        int oc = mhalf * 64 + mt * 16 + lane15;
        int dw = (kv * CCH + oc) * (CIN / 2) + cb * 16 + hi * 8;
        const uint4* ap = reinterpret_cast<const uint4*>(
            (const unsigned int*)wext16 + dw);
        uint4 a0 = ap[0];
        uint4 a1 = ap[1];
        U16x16 Am;
        Am.u[0] = a0.x; Am.u[1] = a0.y; Am.u[2] = a0.z; Am.u[3] = a0.w;
        Am.u[4] = a1.x; Am.u[5] = a1.y; Am.u[6] = a1.z; Am.u[7] = a1.w;
        acc[mt] = __builtin_amdgcn_wmma_f32_16x16x32_f16(
            false, Am.v, false, Bm.v, (short)0, acc[mt], false, false);
      }
    }
  }

  // Epilogue: + b_ext, pack f32 -> f16 pairs, store pixel-major.
#pragma unroll
  for (int mt = 0; mt < 4; ++mt) {
    int Mbase = mhalf * 64 + mt * 16;
#pragma unroll
    for (int t = 0; t < 4; ++t) {
      int ch = Mbase + hi * 8 + 2 * t;
      float v0 = acc[mt][2 * t]     + bext[ch];
      float v1 = acc[mt][2 * t + 1] + bext[ch + 1];
      U32h u;
      u.b[0] = f32_to_f16b(v0);
      u.b[1] = f32_to_f16b(v1);
      fws[(((size_t)b * (CCH / 2) + (ch >> 1)) * HH + i) * WW + px] = u.u;
    }
  }
}

// ---------------------------------------------------------------------------
// Stage 2: 5x5 autocorrelation + contraction, correlation product fused into
// the WMMA B-operand build (v_pk_mul_f16).  LDS tile [cp=64][5 rows][72 cols],
// interior cols 4..67 async-copied, halos zeroed.
// ---------------------------------------------------------------------------
#define FCOLS 72
__global__ void __launch_bounds__(128)
autocorr_wmma(const unsigned int* __restrict__ fws,
              const unsigned short* __restrict__ wreg16,
              const float* __restrict__ breg,
              float* __restrict__ out) {
  __shared__ unsigned int lds[(CCH / 2) * 5 * FCOLS]; // 23040 dwords = 90 KB

  const int wg = blockIdx.x;
  const int b  = wg >> 6;
  const int i  = wg & 63;
  const int tid = threadIdx.x;

  const uint4 z4 = {0u, 0u, 0u, 0u};
  const int NSEG = (CCH / 2) * 5;                 // 320 (cp, row) segments
  for (int s = tid; s < NSEG; s += 128) {
    *(uint4*)&lds[s * FCOLS + 0]  = z4;
    *(uint4*)&lds[s * FCOLS + 68] = z4;
  }
  for (int w = tid; w < NSEG * 16; w += 128) {
    int s  = w >> 4, q = w & 15;
    int cp = s / 5, r = s % 5;
    int gi = i + r - 2;
    int lidx = s * FCOLS + 4 + q * 4;
    if (gi >= 0 && gi < HH) {
      const unsigned int* gp =
          fws + ((((size_t)b * (CCH / 2) + cp) * HH + gi) * WW + q * 4);
      async_g2l_b128((unsigned)(uintptr_t)&lds[lidx], gp);
    } else {
      *(uint4*)&lds[lidx] = z4;
    }
  }
  wait_asynccnt0();
  __syncthreads();

  const int lane   = tid & 31;
  const int wv     = tid >> 5;
  const int lane15 = lane & 15;
  const int hi     = (lane >> 4) & 1;
  const int px     = wv * 16 + lane15;

  v8f acc[4] = {};

  for (int cb = 0; cb < 4; ++cb) {           // 4 blocks of 32 channels
    h2 cen[8];
#pragma unroll
    for (int jv = 0; jv < 8; ++jv) {         // hoist center pairs (u=v=2 -> col j+4)
      int cp = cb * 16 + jv + hi * 8;
      U32h u;
      u.u = lds[(cp * 5 + 2) * FCOLS + (px + 4)];
      cen[jv] = u.h;
    }
    for (int k = 0; k < 25; ++k) {
      const int u5 = k / 5, v5 = k % 5;
      const int col = px + v5 + 2;           // gj = j+v-2 -> col = gj+4
      U16x16 Bm;
#pragma unroll
      for (int jv = 0; jv < 8; ++jv) {
        int cp = cb * 16 + jv + hi * 8;
        U32h s;
        s.u = lds[(cp * 5 + u5) * FCOLS + col];
        U32h pr;
        pr.h = cen[jv] * s.h;                // v_pk_mul_f16: corr = center*shift
        Bm.u[jv] = pr.u;
      }
#pragma unroll
      for (int mt = 0; mt < 4; ++mt) {
        int oc = mt * 16 + lane15;
        int dw = (k * COUTC + oc) * (CCH / 2) + cb * 16 + hi * 8;
        const uint4* ap = reinterpret_cast<const uint4*>(
            (const unsigned int*)wreg16 + dw);
        uint4 a0 = ap[0];
        uint4 a1 = ap[1];
        U16x16 Am;
        Am.u[0] = a0.x; Am.u[1] = a0.y; Am.u[2] = a0.z; Am.u[3] = a0.w;
        Am.u[4] = a1.x; Am.u[5] = a1.y; Am.u[6] = a1.z; Am.u[7] = a1.w;
        acc[mt] = __builtin_amdgcn_wmma_f32_16x16x32_f16(
            false, Am.v, false, Bm.v, (short)0, acc[mt], false, false);
      }
    }
  }

  // Epilogue: + b_reg, write NCHW f32.
#pragma unroll
  for (int mt = 0; mt < 4; ++mt) {
#pragma unroll
    for (int r = 0; r < 8; ++r) {
      int oc = mt * 16 + hi * 8 + r;
      out[(((size_t)b * COUTC + oc) * HH + i) * WW + px] = acc[mt][r] + breg[oc];
    }
  }
}

// ---------------------------------------------------------------------------
// Workspace map (bytes):
//   [0,            589824)   wext16 : 9*128*256 f16
//   [589824,       999424)   wreg16 : 25*64*128 f16
//   [999424,      9388032)   fws    : 8*64*64*64 packed f16 pairs (u32)
//   [9388032,    26165248)   xp     : 8*128*64*64 packed f16 pairs (u32)
// ---------------------------------------------------------------------------
extern "C" void kernel_launch(void* const* d_in, const int* in_sizes, int n_in,
                              void* d_out, int out_size, void* d_ws, size_t ws_size,
                              hipStream_t stream) {
  const float* x    = (const float*)d_in[0];
  const float* wext = (const float*)d_in[1];
  const float* bext = (const float*)d_in[2];
  const float* wreg = (const float*)d_in[3];
  const float* breg = (const float*)d_in[4];

  char* ws = (char*)d_ws;
  unsigned short* wext16 = (unsigned short*)(ws);
  unsigned short* wreg16 = (unsigned short*)(ws + 589824);
  unsigned int*   fws    = (unsigned int*)(ws + 999424);
  unsigned int*   xp     = (unsigned int*)(ws + 9388032);

  prep_convert<<<2048, 256, 0, stream>>>(wext, wreg, x, wext16, wreg16, xp);
  conv3x3_wmma<<<BATCH * HH, 256, 0, stream>>>(xp, wext16, bext, fws);
  autocorr_wmma<<<BATCH * HH, 128, 0, stream>>>(fws, wreg16, breg, (float*)d_out);
}
